// DynamicWeightsCat33_28097676050579
// MI455X (gfx1250) — matrix operand order repaired and checked by reading, not compiled, not run on hardware
//
#include <hip/hip_runtime.h>

typedef __attribute__((ext_vector_type(16))) __bf16 v16bf;
typedef __attribute__((ext_vector_type(8)))  float  v8f;

namespace {
constexpr int N_   = 2;
constexpr int C_   = 256;
constexpr int H_   = 96;
constexpr int W_   = 96;
constexpr int P_   = H_ * W_;          // 9216 pixels per image
constexpr int K9   = 9;                // 3x3 taps
constexpr int KDIM = C_ * K9;          // 2304, K of the logits GEMM
constexpr int KS1  = KDIM / 32;        // 72 WMMA k-steps (bf16 k=32)
constexpr int CAT  = 5 * C_;           // 1280, K of the projection GEMM
constexpr int KS3  = CAT / 32;         // 40 WMMA k-steps
constexpr int TPR  = P_ / 16;          // 576 16-pixel tiles per image (16 | 96)
constexpr int T64  = P_ / 64;          // 144 64-pixel tiles per image
}

static __device__ __forceinline__ unsigned short f2bf(float f) {
    unsigned int u = __float_as_uint(f);
    return (unsigned short)((u + 0x7fffu + ((u >> 16) & 1u)) >> 16);  // RNE
}
static __device__ __forceinline__ float bf2f(unsigned short h) {
    return __uint_as_float(((unsigned int)h) << 16);
}

union Frag16 { v16bf v; unsigned short u[16]; };

// ---------------- packing kernels ----------------

// branch conv weights -> WMMA A-fragment order: A[bi][ks][lane][e], 32B per lane
// A packing used here (16x32, contiguous k, same order as B): lane L: M=L%16, half=L/16;
// element e: K = half*16 + e.  K ordered tap-major: K = t*256 + c.
__global__ void k_pack_brA(const float* __restrict__ wa, const float* __restrict__ wb,
                           const float* __restrict__ wc, const float* __restrict__ wd,
                           unsigned short* __restrict__ A) {
    int tid = blockIdx.x * 256 + threadIdx.x;            // 4*KS1*32*16 = 147456
    if (tid >= 4 * KS1 * 512) return;
    int e    = tid & 15;
    int lane = (tid >> 4) & 31;
    int ks   = (tid >> 9) % KS1;
    int bi   = (tid >> 9) / KS1;
    int M    = lane & 15, half = lane >> 4;
    int K    = ks * 32 + half * 16 + e;
    int t    = K >> 8;            // tap 0..8
    int c    = K & 255;           // channel
    const float* w = (bi == 0) ? wa : (bi == 1) ? wb : (bi == 2) ? wc : wd;
    float v = (M < K9) ? w[(M * C_ + c) * K9 + t] : 0.f;
    A[tid] = f2bf(v);
}

// w_proj [256][1280] -> WMMA A-fragment order: A[mt][ks][lane][e]
__global__ void k_pack_prA(const float* __restrict__ wp, unsigned short* __restrict__ A) {
    int tid = blockIdx.x * 256 + threadIdx.x;            // 16*KS3*32*16 = 327680
    if (tid >= 16 * KS3 * 512) return;
    int e    = tid & 15;
    int lane = (tid >> 4) & 31;
    int ks   = (tid >> 9) % KS3;
    int mt   = (tid >> 9) / KS3;
    int M    = mt * 16 + (lane & 15);
    int half = lane >> 4;
    int K    = ks * 32 + half * 16 + e;
    A[tid] = f2bf(wp[M * CAT + K]);
}

// feat[n][p][0:256] = x[n][:,p]  (pixel-major bf16 "cat" buffer, coalesced writes)
__global__ void k_xfeat(const float* __restrict__ x, unsigned short* __restrict__ feat) {
    int tid = blockIdx.x * 256 + threadIdx.x;
    if (tid >= N_ * P_ * C_) return;
    int c  = tid & 255;
    int np = tid >> 8;               // n*P_ + p
    int n  = np / P_;
    int p  = np - n * P_;
    feat[(size_t)np * CAT + c] = f2bf(x[((size_t)n * C_ + c) * P_ + p]);
}

// ---------------- branch kernel: WMMA logits + softmax + dynamic filter ----------------
// Reads the x-section of feat (pixel-major), writes the branch section of feat.

__global__ __launch_bounds__(32)
void k_branch(unsigned short* feat,
              const unsigned short* __restrict__ Abr) {
    int wg   = blockIdx.x;                 // N_*4*TPR = 4608
    int tile = wg % TPR;
    int bi   = (wg / TPR) & 3;
    int n    = wg / (TPR * 4);
    int d    = (bi == 0) ? 1 : bi * 4;     // 1,4,8,12
    int p0   = tile * 16;
    int h0   = p0 / W_;
    int w0   = p0 - h0 * W_;               // whole tile sits on one row (16 | 96)

    int lane = threadIdx.x;
    int Nn   = lane & 15;
    int half = lane >> 4;

    const unsigned short* Apk   = Abr + (size_t)bi * (KS1 * 512);
    unsigned short*       featn = feat + (size_t)n * P_ * CAT;

    Frag16 zf;
#pragma unroll
    for (int e = 0; e < 16; ++e) zf.u[e] = 0;

    v8f acc = {};
    // K ordered tap-major; each tap covers 8 k-steps (256 channels / 32).
    // Within a tap block the shift (and bounds check) is uniform for both lane halves.
#pragma unroll 1
    for (int t = 0; t < K9; ++t) {
        int dy = (t / 3 - 1) * d, dx = (t % 3 - 1) * d;
        int hh = h0 + dy, ww = w0 + Nn + dx;
        bool ok = (hh >= 0) && (hh < H_) && (ww >= 0) && (ww < W_);
        int pix = hh * W_ + ww;            // valid only when ok
        const unsigned short* Apt = Apk + (size_t)(t * 8) * 512;
#pragma unroll 1
        for (int k8 = 0; k8 < 8; ++k8) {
            Frag16 a;
            a.v = *(const v16bf*)(Apt + (size_t)(k8 * 32 + lane) * 16);
            int c0 = k8 * 32 + half * 16;  // 16 consecutive channels
            v16bf b;
            if (ok) b = *(const v16bf*)(featn + (size_t)pix * CAT + c0);
            else    b = zf.v;
            acc = __builtin_amdgcn_wmma_f32_16x16x32_bf16(false, a.v, false, b,
                                                          (short)0, acc, false, false);
        }
    }

    // acc: lane column Nn; half0 lanes hold logit rows 0..7, half1 lane element 0 = row 8
    __shared__ float fsh[K9][16];
    float mh;
    if (half == 0) {
        mh = acc[0];
#pragma unroll
        for (int v = 1; v < 8; ++v) mh = fmaxf(mh, acc[v]);
    } else {
        mh = acc[0];
    }
    float mx = fmaxf(mh, __shfl_xor(mh, 16, 32));
    float sh;
    if (half == 0) {
        sh = 0.f;
#pragma unroll
        for (int v = 0; v < 8; ++v) sh += expf(acc[v] - mx);
    } else {
        sh = expf(acc[0] - mx);
    }
    float inv = 1.f / (sh + __shfl_xor(sh, 16, 32));
    if (half == 0) {
#pragma unroll
        for (int v = 0; v < 8; ++v) fsh[v][Nn] = expf(acc[v] - mx) * inv;
    } else {
        fsh[8][Nn] = expf(acc[0] - mx) * inv;
    }
    __syncthreads();

    // dynamic filter: out[c][p] = sum_t fsh[t][p] * x[c][p shifted by tap t]
    // lane = channel -> loads and stores are contiguous 64B across the wave
#pragma unroll 1
    for (int cb = 0; cb < 8; ++cb) {
        int c = cb * 32 + lane;
        float o[16];
#pragma unroll
        for (int pp = 0; pp < 16; ++pp) o[pp] = 0.f;
#pragma unroll
        for (int t = 0; t < K9; ++t) {
            int dy = (t / 3 - 1) * d, dx = (t % 3 - 1) * d;
            int hh  = h0 + dy;
            bool rok = (hh >= 0 && hh < H_);
            int rowb = hh * W_;
#pragma unroll
            for (int pp = 0; pp < 16; ++pp) {
                int ww = w0 + pp + dx;
                float xv = (rok && ww >= 0 && ww < W_)
                           ? bf2f(featn[(size_t)(rowb + ww) * CAT + c]) : 0.f;
                o[pp] = fmaf(fsh[t][pp], xv, o[pp]);
            }
        }
#pragma unroll
        for (int pp = 0; pp < 16; ++pp)
            featn[(size_t)(p0 + pp) * CAT + C_ + bi * C_ + c] = f2bf(o[pp]);
    }
}

// ---------------- 1x1 projection GEMM (WMMA) ----------------

__global__ __launch_bounds__(32)
void k_proj(const unsigned short* __restrict__ feat,
            const unsigned short* __restrict__ Apr,
            const float* __restrict__ bias,
            float* __restrict__ y) {
    int wg = blockIdx.x;                   // N_*16*T64 = 4608
    int pt = wg % T64;
    int mt = (wg / T64) & 15;
    int n  = wg / (T64 * 16);
    int p0 = pt * 64;

    int lane = threadIdx.x;
    int Nn   = lane & 15;
    int half = lane >> 4;

    const unsigned short* Apk = Apr + (size_t)mt * (KS3 * 512);
    const unsigned short* fb  = feat + (size_t)n * P_ * CAT;
    const unsigned short* fbs[4];
#pragma unroll
    for (int s = 0; s < 4; ++s)
        fbs[s] = fb + (size_t)(p0 + s * 16 + Nn) * CAT + half * 16;

    v8f acc[4] = {};
#pragma unroll 1
    for (int ks = 0; ks < KS3; ++ks) {
        Frag16 a;
        a.v = *(const v16bf*)(Apk + (size_t)(ks * 32 + lane) * 16);
        if (ks + 1 < KS3)
            __builtin_prefetch(Apk + (size_t)((ks + 1) * 32 + lane) * 16, 0, 1);
#pragma unroll
        for (int s = 0; s < 4; ++s) {
            v16bf b = *(const v16bf*)(fbs[s] + ks * 32);   // contiguous 32B, K=half*16+e
            acc[s] = __builtin_amdgcn_wmma_f32_16x16x32_bf16(false, a.v, false, b,
                                                             (short)0, acc[s], false, false);
        }
    }
#pragma unroll
    for (int s = 0; s < 4; ++s) {
        int p = p0 + s * 16 + Nn;
#pragma unroll
        for (int v = 0; v < 8; ++v) {
            int co = mt * 16 + half * 8 + v;               // C/D row M = half*8+v
            y[((size_t)n * C_ + co) * P_ + p] = acc[s][v] + bias[co];
        }
    }
}

// ---------------- GroupNorm + ReLU ----------------

__global__ void k_gn_stats(const float* __restrict__ y, float* __restrict__ st) {
    int b   = blockIdx.x;                  // n*32 + g, 64 total; channels b*8..b*8+7
    int tid = threadIdx.x;
    const float* base = y + (size_t)b * 8 * P_;
    const int count = 8 * P_;              // 73728
    float s = 0.f, ss = 0.f;
    for (int i = tid; i < count; i += 256) {
        float v = base[i];
        s += v; ss += v * v;
    }
    __shared__ float rs[256], rq[256];
    rs[tid] = s; rq[tid] = ss;
    __syncthreads();
    for (int o = 128; o > 0; o >>= 1) {
        if (tid < o) { rs[tid] += rs[tid + o]; rq[tid] += rq[tid + o]; }
        __syncthreads();
    }
    if (tid == 0) {
        float mu  = rs[0] / (float)count;
        float var = rq[0] / (float)count - mu * mu;
        st[b * 2]     = mu;
        st[b * 2 + 1] = rsqrtf(var + 1e-5f);
    }
}

__global__ void k_gn_apply(const float* __restrict__ y, const float* __restrict__ st,
                           const float* __restrict__ gamma, const float* __restrict__ beta,
                           float* __restrict__ out) {
    int tid = blockIdx.x * 256 + threadIdx.x;
    if (tid >= N_ * C_ * P_) return;
    int cp = tid / P_;
    int c  = cp & 255;
    int n  = cp >> 8;
    int g  = c >> 3;
    int sb = (n * 32 + g) * 2;
    float v = (y[tid] - st[sb]) * st[sb + 1] * gamma[c] + beta[c];
    out[tid] = fmaxf(v, 0.f);
}

// ---------------- launcher ----------------

extern "C" void kernel_launch(void* const* d_in, const int* in_sizes, int n_in,
                              void* d_out, int out_size, void* d_ws, size_t ws_size,
                              hipStream_t stream) {
    (void)in_sizes; (void)n_in; (void)out_size; (void)ws_size;
    const float* x     = (const float*)d_in[0];
    const float* wa    = (const float*)d_in[1];
    const float* wb    = (const float*)d_in[2];
    const float* wc    = (const float*)d_in[3];
    const float* wd    = (const float*)d_in[4];
    const float* wp    = (const float*)d_in[5];
    const float* bproj = (const float*)d_in[6];
    const float* gam   = (const float*)d_in[7];
    const float* bet   = (const float*)d_in[8];
    float* out = (float*)d_out;

    unsigned char* ws = (unsigned char*)d_ws;
    size_t off = 0;
    unsigned short* feat = (unsigned short*)(ws + off); off += (size_t)N_ * P_ * CAT * 2;   // 47.19 MB
    unsigned short* Abr  = (unsigned short*)(ws + off); off += (size_t)4 * KS1 * 512 * 2;   //  0.29 MB
    unsigned short* Apr  = (unsigned short*)(ws + off); off += (size_t)16 * KS3 * 512 * 2;  //  0.66 MB
    float*          y    = (float*)(ws + off);          off += (size_t)N_ * C_ * P_ * 4;    // 18.87 MB
    float*          st   = (float*)(ws + off);                                              //  512 B

    k_pack_brA<<<(4 * KS1 * 512 + 255) / 256, 256, 0, stream>>>(wa, wb, wc, wd, Abr);
    k_pack_prA<<<(16 * KS3 * 512 + 255) / 256, 256, 0, stream>>>(wp, Apr);
    k_xfeat   <<<(N_ * P_ * C_ + 255) / 256, 256, 0, stream>>>(x, feat);
    k_branch  <<<N_ * 4 * TPR, 32, 0, stream>>>(feat, Abr);
    k_proj    <<<N_ * 16 * T64, 32, 0, stream>>>(feat, Apr, bproj, y);
    k_gn_stats<<<2 * 32, 256, 0, stream>>>(y, st);
    k_gn_apply<<<(N_ * C_ * P_ + 255) / 256, 256, 0, stream>>>(y, st, gam, bet, out);
}
